// NodeClassifier_56221121904890
// MI455X (gfx1250) — matrix operand order repaired and checked by reading, not compiled, not run on hardware
//
#include <hip/hip_runtime.h>
#include <hip/hip_bf16.h>

// ---------------------------------------------------------------------------
// GraphSAGE (3 layers) + BN + linear head for MI455X (gfx1250, wave32, WMMA).
//
// - Message Linear hoisted from per-edge to per-node (message depends only on
//   the source node): GEMM work drops from ~52 GFLOP to ~4.5 GFLOP.
// - Working set (~100 MB) is L2-resident (192 MB), so the edge scatter phase
//   is bound by L2 gather traffic + atomic throughput. Message features y are
//   stored as bf16 (they come out of a bf16 WMMA anyway), halving the
//   per-edge gather to 128 B; the mean accumulation stays in f32 atomics.
// - GEMMs use v_wmma_f32_16x16x32_bf16, f32 accumulate, K compile-time so the
//   K loop fully unrolls (2/4/6 WMMAs per instantiation).
// ---------------------------------------------------------------------------

typedef __attribute__((ext_vector_type(16))) __bf16 v16bf;
typedef __attribute__((ext_vector_type(8)))  float  v8f;

union Frag16 {
    unsigned short us[16];
    uint4          q[2];
    v16bf          bf;
};

// f32 -> bf16 with round-to-nearest-even
__device__ __forceinline__ unsigned short f2bf(float f) {
    unsigned int u = __float_as_uint(f);
    u += 0x7FFFu + ((u >> 16) & 1u);
    return (unsigned short)(u >> 16);
}

// ---------------------------------------------------------------------------
// C[N,64] = relu( concat(A1[N,K1], A2[N,K2]) @ W[64,K]^T (+ bias) )
// Block: 256 threads = 8 waves; block covers 32 rows x 64 cols
//   wave&1 -> M tile (2 x 16 rows), wave>>1 -> N tile (4 x 16 cols)
// W staged in LDS as bf16, row stride K+8 (16B-aligned rows, no bank
// conflicts). K1, K2 multiples of 8; K multiple of 32 -> full unroll.
// BF16_OUT: store result as packed bf16 rows (for the edge-gather path).
// ---------------------------------------------------------------------------
template <int K1, int K2, bool HAS_BIAS, bool BF16_OUT>
__global__ void __launch_bounds__(256)
gemm_relu_h64(const float* __restrict__ A1,
              const float* __restrict__ A2,
              const float* __restrict__ W,
              const float* __restrict__ bias,
              void* __restrict__ out, int N)
{
    constexpr int K  = K1 + K2;
    constexpr int KP = K + 8;
    __shared__ unsigned short smemW[64 * KP];

    // Stage W ([64,K] f32, row-major) into LDS as bf16.
    #pragma unroll 2
    for (int idx = threadIdx.x; idx < 64 * K; idx += 256) {
        int r = idx / K;          // compile-time K -> strength-reduced
        int c = idx - r * K;
        smemW[r * KP + c] = f2bf(W[r * K + c]);
    }
    __syncthreads();

    const int wave = threadIdx.x >> 5;
    const int lane = threadIdx.x & 31;
    const int m0 = blockIdx.x * 32 + ((wave & 1) << 4);
    const int n0 = (wave >> 1) << 4;
    const int lr = lane & 15;   // A: row-in-tile, B/C: col-in-tile
    const int lh = lane >> 4;   // half select

    int arow = m0 + lr;
    if (arow >= N) arow = N - 1;   // clamp (stores guarded; keeps EXEC full)

    // Prefetch A rows (gfx1250 global_prefetch_b8).
    __builtin_prefetch(A1 + (size_t)arow * K1, 0, 3);
    if (K2 > 0) __builtin_prefetch(A2 + (size_t)arow * K2, 0, 3);

    v8f acc = {};

    #pragma unroll
    for (int k0 = 0; k0 < K; k0 += 32) {
        // ---- A fragment: 16x32 bf16 (ISA 7.12.2):
        //   lanes 0-15 : K = k0+0..7  and k0+16..23   (M = lane)
        //   lanes 16-31: K = k0+8..15 and k0+24..31   (M = lane-16)
        Frag16 a;
        const int kA = k0 + (lh << 3);
        #pragma unroll
        for (int half = 0; half < 2; ++half) {
            const int kk = kA + half * 16;   // always < K
            // K1 % 8 == 0: an 8-chunk never straddles the concat boundary.
            const float* p;
            if (k0 + 32 <= K1) {
                p = A1 + (size_t)arow * K1 + kk;
            } else if (k0 >= K1) {
                p = A2 + (size_t)arow * K2 + (kk - K1);
            } else {
                p = (kk < K1) ? (A1 + (size_t)arow * K1 + kk)
                              : (A2 + (size_t)arow * K2 + (kk - K1));
            }
            float4 u = *(const float4*)(p);
            float4 v = *(const float4*)(p + 4);
            const int o = half * 8;
            a.us[o + 0] = f2bf(u.x); a.us[o + 1] = f2bf(u.y);
            a.us[o + 2] = f2bf(u.z); a.us[o + 3] = f2bf(u.w);
            a.us[o + 4] = f2bf(v.x); a.us[o + 5] = f2bf(v.y);
            a.us[o + 6] = f2bf(v.z); a.us[o + 7] = f2bf(v.w);
        }

        // ---- B fragment: 32x16 bf16 (B = W^T, B[k][n] = W[n][k]).
        //   lanes 0-15 : K = k0+0..7 and k0+16..23 (N = lane); lanes 16-31: K = k0+8..15 and k0+24..31
        // two 8-bf16 chunks of W row (n0+lr) -> two ds_load_b128.
        Frag16 b;
        const unsigned short* wp = &smemW[(n0 + lr) * KP + k0 + (lh << 3)];
        b.q[0] = *(const uint4*)(wp);
        b.q[1] = *(const uint4*)(wp + 16);

        acc = __builtin_amdgcn_wmma_f32_16x16x32_bf16(
            /*neg_a=*/false, a.bf, /*neg_b=*/false, b.bf,
            /*c_mod=*/(short)0, acc, /*reuse_a=*/false, /*reuse_b=*/false);
    }

    // Epilogue. C layout: lane = N, VGPR j = M (+8 for upper half-lanes).
    const int   ncol  = n0 + lr;
    const float bv    = HAS_BIAS ? bias[ncol] : 0.0f;
    const int   mbase = m0 + (lh << 3);
    const bool  full  = (m0 + 16 <= N);   // wave-uniform -> scalar branch
    if constexpr (BF16_OUT) {
        unsigned short* op = (unsigned short*)out + (size_t)mbase * 64 + ncol;
        if (full) {
            #pragma unroll
            for (int j = 0; j < 8; ++j)
                op[(size_t)j * 64] = f2bf(fmaxf(acc[j] + bv, 0.0f));
        } else {
            #pragma unroll
            for (int j = 0; j < 8; ++j)
                if (mbase + j < N)
                    op[(size_t)j * 64] = f2bf(fmaxf(acc[j] + bv, 0.0f));
        }
    } else {
        float* op = (float*)out + (size_t)mbase * 64 + ncol;
        if (full) {
            #pragma unroll
            for (int j = 0; j < 8; ++j)
                op[(size_t)j * 64] = fmaxf(acc[j] + bv, 0.0f);
        } else {
            #pragma unroll
            for (int j = 0; j < 8; ++j)
                if (mbase + j < N)
                    op[(size_t)j * 64] = fmaxf(acc[j] + bv, 0.0f);
        }
    }
}

// ---------------------------------------------------------------------------
// Edge scatter: sums[col] += bf16_unpack(y16[row]) (64 features).
// 32 consecutive lanes handle one edge row: each lane loads one packed u32
// (two bf16 features) and issues two f32 global atomics. Gather is half the
// bytes of an f32 layout; accumulation precision stays f32.
// ---------------------------------------------------------------------------
__global__ void __launch_bounds__(256)
scatter_accum(const unsigned int* __restrict__ y16,
              const long long* __restrict__ row,
              const long long* __restrict__ col,
              float* __restrict__ sums, long long EH2)
{
    long long idx = (long long)blockIdx.x * 256 + threadIdx.x;
    if (idx >= EH2) return;
    long long e = idx >> 5;
    int f2 = (int)(idx & 31);                 // feature pair index
    int r = (int)row[e];
    int c = (int)col[e];
    unsigned int p = y16[(size_t)r * 32 + f2];
    float lo = __uint_as_float(p << 16);          // feature 2*f2
    float hi = __uint_as_float(p & 0xFFFF0000u);  // feature 2*f2+1
    float* d = &sums[(size_t)c * 64 + 2 * f2];
    atomicAdd(d,     lo);
    atomicAdd(d + 1, hi);
}

// Degree counts, one thread per edge.
__global__ void __launch_bounds__(256)
degree_count(const long long* __restrict__ col, float* __restrict__ cnt,
             long long E)
{
    long long e = (long long)blockIdx.x * 256 + threadIdx.x;
    if (e >= E) return;
    atomicAdd(&cnt[(int)col[e]], 1.0f);
}

__global__ void __launch_bounds__(256)
mean_div(float* __restrict__ sums, const float* __restrict__ cnt, int NH)
{
    int idx = blockIdx.x * 256 + threadIdx.x;
    if (idx >= NH) return;
    float c = cnt[idx >> 6];
    sums[idx] *= 1.0f / fmaxf(c, 1.0f);
}

// ---------------------------------------------------------------------------
// BatchNorm stats: stats[0:64] = sum, stats[64:128] = sum of squares.
// ---------------------------------------------------------------------------
__global__ void __launch_bounds__(256)
bn_reduce(const float* __restrict__ h, float* __restrict__ stats, int N)
{
    __shared__ float ss[128];
    int tid = threadIdx.x;
    if (tid < 128) ss[tid] = 0.0f;
    __syncthreads();
    int f   = tid & 63;
    int sub = tid >> 6;                       // 0..3
    int r0   = blockIdx.x * 1024 + sub;
    int rend = (blockIdx.x + 1) * 1024;
    if (rend > N) rend = N;
    float s = 0.0f, q = 0.0f;
    for (int r = r0; r < rend; r += 4) {
        float v = h[(size_t)r * 64 + f];
        s += v; q += v * v;
    }
    atomicAdd(&ss[f], s);
    atomicAdd(&ss[64 + f], q);
    __syncthreads();
    if (tid < 128) atomicAdd(&stats[tid], ss[tid]);
}

__global__ void __launch_bounds__(256)
bn_apply_relu(float* __restrict__ h, const float* __restrict__ stats,
              const float* __restrict__ gamma, const float* __restrict__ beta,
              int N)
{
    int idx = blockIdx.x * 256 + threadIdx.x;
    if (idx >= N * 64) return;
    int f = idx & 63;
    float invN = 1.0f / (float)N;
    float mu   = stats[f] * invN;
    float var  = stats[64 + f] * invN - mu * mu;
    float sc   = rsqrtf(fmaxf(var, 0.0f) + 1e-5f) * gamma[f];
    float v    = (h[idx] - mu) * sc + beta[f];
    h[idx]     = fmaxf(v, 0.0f);
}

// ---------------------------------------------------------------------------
// Head: out[N,10] = h[N,64] @ W[10,64]^T + b. 16 lanes per node (6 idle).
// ---------------------------------------------------------------------------
__global__ void __launch_bounds__(256)
final_linear(const float* __restrict__ h, const float* __restrict__ W,
             const float* __restrict__ b, float* __restrict__ out, int N)
{
    __shared__ float sw[10 * 64];
    __shared__ float sb[10];
    for (int i = threadIdx.x; i < 640; i += 256) sw[i] = W[i];
    if (threadIdx.x < 10) sb[threadIdx.x] = b[threadIdx.x];
    __syncthreads();
    long long idx = (long long)blockIdx.x * 256 + threadIdx.x;
    long long n = idx >> 4;
    int o = (int)(idx & 15);
    if (n >= N || o >= 10) return;
    const float* hr = h + n * 64;
    const float* wr = sw + o * 64;
    float acc = sb[o];
    #pragma unroll
    for (int k = 0; k < 64; ++k) acc = fmaf(hr[k], wr[k], acc);
    out[n * 10 + o] = acc;
}

// ---------------------------------------------------------------------------
extern "C" void kernel_launch(void* const* d_in, const int* in_sizes, int n_in,
                              void* d_out, int out_size, void* d_ws, size_t ws_size,
                              hipStream_t stream)
{
    const float*     x  = (const float*)d_in[0];
    const long long* ei = (const long long*)d_in[1];
    // d_in[2] = pos, unused by the reference network
    const float* lin1_w = (const float*)d_in[3];
    const float* lin1_b = (const float*)d_in[4];
    const float* upd1_w = (const float*)d_in[5];
    const float* lin2_w = (const float*)d_in[6];
    const float* lin2_b = (const float*)d_in[7];
    const float* upd2_w = (const float*)d_in[8];
    const float* lin3_w = (const float*)d_in[9];
    const float* lin3_b = (const float*)d_in[10];
    const float* upd3_w = (const float*)d_in[11];
    const float* bn_g   = (const float*)d_in[12];
    const float* bn_b   = (const float*)d_in[13];
    const float* out_w  = (const float*)d_in[14];
    const float* out_b  = (const float*)d_in[15];

    const int       N = in_sizes[2] / 3;      // pos is [N,3]
    const long long E = in_sizes[1] / 2;      // edge_index is [2,E]
    const long long* row = ei;
    const long long* col = ei + E;

    // Workspace layout (float units):
    const size_t nh = (size_t)N * 64;
    float* ws    = (float*)d_ws;
    float* yf    = ws;                 // [N,32 u32] bf16-packed node messages
    float* sums  = yf + nh / 2;        // [N,64] f32 scatter accum / aggr
    float* cnt   = sums + nh;          // [N]    degree counts
    float* hA    = cnt + (((size_t)N + 255) & ~(size_t)255);
    float* hB    = hA + nh;
    float* stats = hB + nh;            // [128] BN sum/sumsq
    unsigned int* y16 = (unsigned int*)yf;

    dim3 blk(256);
    const int       gemm_blocks = (N + 31) / 32;
    const long long EH2         = E * 32;     // feature-pairs per edge
    const int       sc_blocks   = (int)((EH2 + 255) / 256);
    const int       e_blocks    = (int)((E + 255) / 256);
    const int       nh_blocks   = (int)((nh + 255) / 256);
    const int       bn_blocks   = (N + 1023) / 1024;
    const int       fl_blocks   = (int)(((size_t)N * 16 + 255) / 256);

    auto aggregate = [&]() {
        hipMemsetAsync(sums, 0, (nh + (size_t)N) * sizeof(float), stream);
        scatter_accum<<<sc_blocks, blk, 0, stream>>>(y16, row, col, sums, EH2);
        degree_count<<<e_blocks, blk, 0, stream>>>(col, cnt, E);
        mean_div<<<nh_blocks, blk, 0, stream>>>(sums, cnt, (int)nh);
    };
    auto batchnorm = [&](float* h) {
        hipMemsetAsync(stats, 0, 128 * sizeof(float), stream);
        bn_reduce<<<bn_blocks, blk, 0, stream>>>(h, stats, N);
        bn_apply_relu<<<nh_blocks, blk, 0, stream>>>(h, stats, bn_g, bn_b, N);
    };

    // ---- Layer 1: x[N,128] -> hA[N,64], then BN+ReLU
    gemm_relu_h64<128, 0, true, true><<<gemm_blocks, blk, 0, stream>>>(
        x, nullptr, lin1_w, lin1_b, y16, N);
    aggregate();
    gemm_relu_h64<64, 128, false, false><<<gemm_blocks, blk, 0, stream>>>(
        sums, x, upd1_w, nullptr, hA, N);
    batchnorm(hA);

    // ---- Layer 2: hA -> hB, then BN+ReLU
    gemm_relu_h64<64, 0, true, true><<<gemm_blocks, blk, 0, stream>>>(
        hA, nullptr, lin2_w, lin2_b, y16, N);
    aggregate();
    gemm_relu_h64<64, 64, false, false><<<gemm_blocks, blk, 0, stream>>>(
        sums, hA, upd2_w, nullptr, hB, N);
    batchnorm(hB);

    // ---- Layer 3: hB -> hA (no BN)
    gemm_relu_h64<64, 0, true, true><<<gemm_blocks, blk, 0, stream>>>(
        hB, nullptr, lin3_w, lin3_b, y16, N);
    aggregate();
    gemm_relu_h64<64, 64, false, false><<<gemm_blocks, blk, 0, stream>>>(
        sums, hB, upd3_w, nullptr, hA, N);

    // ---- Head
    final_linear<<<fl_blocks, blk, 0, stream>>>(hA, out_w, out_b, (float*)d_out, N);

    (void)n_in; (void)out_size; (void)ws_size;
}
